// LTE_74680891343496
// MI455X (gfx1250) — matrix operand order repaired and checked by reading, not compiled, not run on hardware
//
#include <hip/hip_runtime.h>
#include <math.h>

typedef _Float16 half16 __attribute__((ext_vector_type(16)));
typedef float    float8 __attribute__((ext_vector_type(8)));

#define B_   4
#define H_   192
#define W_   192
#define HW_  (H_*W_)
#define Q_   36864
#define HID_ 256
#define HP_  194            // padded spatial extent (zero border for 3x3 convs)
#define PI_F 3.14159265358979323846f

union Frag { uint4 u[2]; half16 h; };

// ---------------------------------------------------------------- zero scratch
__global__ void zero_kernel(uint4* __restrict__ p, long n) {
    long i = (long)blockIdx.x * blockDim.x + threadIdx.x;
    if (i < n) { uint4 z; z.x = z.y = z.z = z.w = 0u; p[i] = z; }
}

// ------------------------------------------------- pack weights into WMMA B-frag order
// Wcf: coef_w||freq_w  -> [kt(3)][nt(32)][lane(32)][e(16)] f16, K padded 81->96
// Wmlp: im_w0..3       -> [layer(4)][kt(8)][nt(16)][lane(32)][e(16)] f16
__global__ void prep_kernel(const float* __restrict__ coefw, const float* __restrict__ freqw,
                            const float* __restrict__ w0, const float* __restrict__ w1,
                            const float* __restrict__ w2, const float* __restrict__ w3,
                            _Float16* __restrict__ wcf, _Float16* __restrict__ wmlp) {
    int tid = blockIdx.x * blockDim.x + threadIdx.x;
    if (tid < 3*32*32*16) {
        int e  = tid & 15;
        int l  = (tid >> 4) & 31;
        int nt = (tid >> 9) & 31;
        int kt = tid >> 14;
        int k  = kt*32 + (e >> 3)*16 + (l >> 4)*8 + (e & 7);
        int n  = nt*16 + (l & 15);
        float v = 0.f;
        if (k < 81) v = (n < 256) ? coefw[n*81 + k] : freqw[(n-256)*81 + k];
        wcf[tid] = (_Float16)v;
    }
    int t2 = tid - 3*32*32*16;
    if (t2 >= 0 && t2 < 4*8*16*32*16) {
        int e     = t2 & 15;
        int l     = (t2 >> 4) & 31;
        int nt    = (t2 >> 9) & 15;
        int kt    = (t2 >> 13) & 7;
        int layer = t2 >> 16;
        int k = kt*32 + (e >> 3)*16 + (l >> 4)*8 + (e & 7);
        int n = nt*16 + (l & 15);
        const float* w = layer == 0 ? w0 : layer == 1 ? w1 : layer == 2 ? w2 : w3;
        wmlp[t2] = (_Float16)w[n*256 + k];
    }
}

// ---------------------------------------------------------- adapter conv1 + sin
// one block (128 threads) per pixel; thread = output channel
__global__ __launch_bounds__(128)
void adapter1_kernel(const float* __restrict__ inp, const float* __restrict__ a1w,
                     const float* __restrict__ a1b, _Float16* __restrict__ t1) {
    __shared__ float s_in[27];
    int pix = blockIdx.x;
    int b = pix / HW_;
    int rem = pix % HW_;
    int y = rem / W_, x = rem % W_;
    int tid = threadIdx.x;
    if (tid < 27) {
        int c = tid / 9, r2 = tid % 9, ky = r2 / 3, kx = r2 % 3;
        int yy = y + ky - 1, xx = x + kx - 1;
        float v = 0.f;
        if (yy >= 0 && yy < H_ && xx >= 0 && xx < W_)
            v = inp[((size_t)(b*3 + c)*H_ + yy)*W_ + xx];
        s_in[tid] = v;
    }
    __syncthreads();
    float acc = a1b[tid];
    #pragma unroll
    for (int j = 0; j < 27; ++j) acc += s_in[j] * a1w[tid*27 + j];
    t1[(((size_t)b*HP_ + (y+1))*HP_ + (x+1))*128 + tid] = (_Float16)sinf(acc);
}

// ----------------------------------- adapter conv2 + sin, build padded feat[...,9] f16
__global__ void adapter2_kernel(const float* __restrict__ inp, const float* __restrict__ score,
                                const float* __restrict__ a2w, const float* __restrict__ a2b,
                                const _Float16* __restrict__ t1, _Float16* __restrict__ feat) {
    int pix = blockIdx.x * blockDim.x + threadIdx.x;
    if (pix >= B_*HW_) return;
    int b = pix / HW_, rem = pix % HW_;
    int y = rem / W_, x = rem % W_;
    float acc0 = a2b[0], acc1 = a2b[1], acc2 = a2b[2];
    for (int ky = 0; ky < 3; ++ky)
        for (int kx = 0; kx < 3; ++kx) {
            const _Float16* tp = t1 + (((size_t)b*HP_ + y + ky)*HP_ + (x + kx))*128;
            int wo = ky*3 + kx;
            for (int i = 0; i < 128; ++i) {
                float v = (float)tp[i];
                acc0 += v * a2w[0*1152 + i*9 + wo];
                acc1 += v * a2w[1*1152 + i*9 + wo];
                acc2 += v * a2w[2*1152 + i*9 + wo];
            }
        }
    _Float16* fb = feat + (((size_t)b*HP_ + (y+1))*HP_ + (x+1))*9;
    #pragma unroll
    for (int c = 0; c < 3; ++c) {
        fb[c]     = (_Float16)inp  [((size_t)(b*3 + c)*H_ + y)*W_ + x];
        fb[3 + c] = (_Float16)score[((size_t)(b*3 + c)*H_ + y)*W_ + x];
    }
    fb[6] = (_Float16)sinf(acc0);
    fb[7] = (_Float16)sinf(acc1);
    fb[8] = (_Float16)sinf(acc2);
}

// ------------------------------------------------------------------ fused query kernel
// one wave per 16-query tile: implicit-GEMM conv (coef||freq, K=96, N=512) via WMMA,
// C staged in LDS, then sinusoidal-encoding epilogue writes X[M,256] f16.
__global__ __launch_bounds__(32)
void lte_query_kernel(const float* __restrict__ coord, const float* __restrict__ cell,
                      const float* __restrict__ coefb, const float* __restrict__ freqb,
                      const float* __restrict__ phasew,
                      const _Float16* __restrict__ feat, const _Float16* __restrict__ wcf,
                      _Float16* __restrict__ act0) {
    __shared__ __align__(16) _Float16 Als[16][96];
    __shared__ float Cls[16][512];
    __shared__ float meta[16][6];

    int lane = threadIdx.x;
    int tile = blockIdx.x;
    int b  = tile / (Q_/16);
    int q0 = (tile % (Q_/16)) * 16;

    // pull the packed conv weights toward this WGP while we do the scalar setup
    __builtin_prefetch(wcf + (size_t)lane*512, 0, 1);

    if (lane < 16) {
        int q = q0 + lane;
        float c0 = coord[((size_t)b*Q_ + q)*2 + 0];
        float c1 = coord[((size_t)b*Q_ + q)*2 + 1];
        const float eps = 1e-6f;
        float cy = fminf(fmaxf(c0, -1.f + eps), 1.f - eps);
        float cx = fminf(fmaxf(c1, -1.f + eps), 1.f - eps);
        float fy = ((cy + 1.f)*H_ - 1.f)*0.5f;
        float fx = ((cx + 1.f)*W_ - 1.f)*0.5f;
        int iy = (int)fminf(fmaxf(rintf(fy), 0.f), (float)(H_-1));
        int ix = (int)fminf(fmaxf(rintf(fx), 0.f), (float)(W_-1));
        float qy = -1.f + (2.f*iy + 1.f)/(float)H_;
        float qx = -1.f + (2.f*ix + 1.f)/(float)W_;
        meta[lane][0] = (c0 - qy)*(float)H_;                       // rel_y
        meta[lane][1] = (c1 - qx)*(float)W_;                       // rel_x
        meta[lane][2] = (float)iy;
        meta[lane][3] = (float)ix;
        meta[lane][4] = cell[((size_t)b*Q_ + q)*2 + 0]*(float)H_;  // rel_cell_y
        meta[lane][5] = cell[((size_t)b*Q_ + q)*2 + 1]*(float)W_;  // rel_cell_x
    }
    __syncthreads();

    // gather 3x3x9 patches into A [16 x 96] (K padded with zeros)
    for (int idx = lane; idx < 16*96; idx += 32) {
        int r = idx / 96, k = idx % 96;
        _Float16 v = (_Float16)0.f;
        if (k < 81) {
            int c = k / 9, rr = k % 9, ky = rr / 3, kx = rr % 3;
            int iy = (int)meta[r][2], ix = (int)meta[r][3];
            v = feat[(((size_t)b*HP_ + iy + ky)*HP_ + (ix + kx))*9 + c];
        }
        Als[r][k] = v;
    }
    __syncthreads();

    int m = lane & 15, g = lane >> 4;
    for (int nt = 0; nt < 32; ++nt) {
        float8 acc = {};
        #pragma unroll
        for (int kt = 0; kt < 3; ++kt) {
            Frag a, bf;
            a.u[0] = *(const uint4*)&Als[m][kt*32 + g*8];
            a.u[1] = *(const uint4*)&Als[m][kt*32 + 16 + g*8];
            const uint4* wp = (const uint4*)(wcf + ((size_t)((kt*32 + nt)*32 + lane))*16);
            bf.u[0] = wp[0]; bf.u[1] = wp[1];
            acc = __builtin_amdgcn_wmma_f32_16x16x32_f16(false, a.h, false, bf.h,
                                                         (short)0, acc, false, false);
        }
        #pragma unroll
        for (int v = 0; v < 8; ++v)
            Cls[v + 8*g][nt*16 + m] = acc[v];
    }
    __syncthreads();

    // epilogue: X[q][j<128]=cos(pi*qf_j)*coef_j ; X[q][j>=128]=sin(pi*qf_{j-128})*coef_j
    for (int i = 0; i < 128; ++i) {
        int idx = lane + 32*i;
        int r = idx >> 8;
        int j = idx & 255;
        int jj = j & 127;
        float f0 = Cls[r][256 + 2*jj]     + freqb[2*jj];
        float f1 = Cls[r][256 + 2*jj + 1] + freqb[2*jj + 1];
        float ph = meta[r][4]*phasew[jj*2 + 0] + meta[r][5]*phasew[jj*2 + 1];
        float qf = f0*meta[r][0] + f1*meta[r][1] + ph;
        float s  = (j < 128) ? cosf(PI_F*qf) : sinf(PI_F*qf);
        float cf = Cls[r][j] + coefb[j];
        act0[((size_t)b*Q_ + q0 + r)*HID_ + j] = (_Float16)(cf * s);
    }
}

// --------------------------------------------------------------- MLP layer (WMMA GEMM)
// one wave per 16-row M tile; all 16 N tiles live in 128 accumulator VGPRs.
__global__ __launch_bounds__(32)
void mlp_kernel(const _Float16* __restrict__ Ain, const _Float16* __restrict__ Wp,
                const float* __restrict__ bias, _Float16* __restrict__ Aout) {
    int lane = threadIdx.x;
    int mt = blockIdx.x;
    int m16 = lane & 15, g = lane >> 4;
    const _Float16* row = Ain + ((size_t)mt*16 + m16)*HID_;

    float8 acc[16];
    #pragma unroll
    for (int nt = 0; nt < 16; ++nt) acc[nt] = (float8){};

    #pragma unroll
    for (int kt = 0; kt < 8; ++kt) {
        // prefetch next K-tile of the packed weight stream (global_prefetch_b8)
        if (kt < 7)
            __builtin_prefetch(Wp + ((size_t)((kt + 1)*16)*32)*16 + (size_t)lane*256, 0, 1);
        Frag a;
        a.u[0] = *(const uint4*)(row + kt*32 + g*8);
        a.u[1] = *(const uint4*)(row + kt*32 + 16 + g*8);
        #pragma unroll
        for (int nt = 0; nt < 16; ++nt) {
            Frag bf;
            const uint4* wp = (const uint4*)(Wp + ((size_t)((kt*16 + nt)*32 + lane))*16);
            bf.u[0] = wp[0]; bf.u[1] = wp[1];
            acc[nt] = __builtin_amdgcn_wmma_f32_16x16x32_f16(false, a.h, false, bf.h,
                                                             (short)0, acc[nt], false, false);
        }
    }

    #pragma unroll
    for (int nt = 0; nt < 16; ++nt) {
        int n = nt*16 + m16;
        float bv = bias[n];
        #pragma unroll
        for (int v = 0; v < 8; ++v) {
            float x = fmaxf(acc[nt][v] + bv, 0.f);
            Aout[((size_t)mt*16 + v + 8*g)*HID_ + n] = (_Float16)x;
        }
    }
}

// ---------------------------------------------- head 256->3 + bilinear(border) residual
__global__ void final_kernel(const _Float16* __restrict__ X, const float* __restrict__ w4,
                             const float* __restrict__ b4, const float* __restrict__ inp,
                             const float* __restrict__ coord, float* __restrict__ out) {
    int gid = blockIdx.x * blockDim.x + threadIdx.x;
    if (gid >= B_*Q_) return;
    int b = gid / Q_;
    const _Float16* xr = X + (size_t)gid*HID_;
    float a0 = b4[0], a1 = b4[1], a2 = b4[2];
    for (int j = 0; j < HID_; ++j) {
        float v = (float)xr[j];
        a0 += v * w4[j];
        a1 += v * w4[256 + j];
        a2 += v * w4[512 + j];
    }
    float cy = coord[(size_t)gid*2 + 0], cx = coord[(size_t)gid*2 + 1];
    float fy = ((cy + 1.f)*H_ - 1.f)*0.5f;
    float fx = ((cx + 1.f)*W_ - 1.f)*0.5f;
    float y0f = floorf(fy), x0f = floorf(fx);
    float wy = fy - y0f, wx = fx - x0f;
    int y0 = (int)fminf(fmaxf(y0f,       0.f), (float)(H_-1));
    int y1 = (int)fminf(fmaxf(y0f + 1.f, 0.f), (float)(H_-1));
    int x0 = (int)fminf(fmaxf(x0f,       0.f), (float)(W_-1));
    int x1 = (int)fminf(fmaxf(x0f + 1.f, 0.f), (float)(W_-1));
    float pred[3] = {a0, a1, a2};
    #pragma unroll
    for (int c = 0; c < 3; ++c) {
        const float* pc = inp + (size_t)(b*3 + c)*H_*W_;
        float v00 = pc[y0*W_ + x0], v01 = pc[y0*W_ + x1];
        float v10 = pc[y1*W_ + x0], v11 = pc[y1*W_ + x1];
        float bil = v00*(1.f-wy)*(1.f-wx) + v01*(1.f-wy)*wx
                  + v10*wy*(1.f-wx)       + v11*wy*wx;
        out[(size_t)gid*3 + c] = pred[c] + bil;
    }
}

// =====================================================================================
extern "C" void kernel_launch(void* const* d_in, const int* in_sizes, int n_in,
                              void* d_out, int out_size, void* d_ws, size_t ws_size,
                              hipStream_t stream) {
    const float* inp    = (const float*)d_in[0];
    const float* score  = (const float*)d_in[1];
    const float* coord  = (const float*)d_in[2];
    const float* cell   = (const float*)d_in[3];
    const float* coefw  = (const float*)d_in[4];
    const float* coefb  = (const float*)d_in[5];
    const float* freqw  = (const float*)d_in[6];
    const float* freqb  = (const float*)d_in[7];
    const float* phasew = (const float*)d_in[8];
    const float* a1w    = (const float*)d_in[9];
    const float* a1b    = (const float*)d_in[10];
    const float* a2w    = (const float*)d_in[11];
    const float* a2b    = (const float*)d_in[12];
    const float* imw0 = (const float*)d_in[13]; const float* imb0 = (const float*)d_in[14];
    const float* imw1 = (const float*)d_in[15]; const float* imb1 = (const float*)d_in[16];
    const float* imw2 = (const float*)d_in[17]; const float* imb2 = (const float*)d_in[18];
    const float* imw3 = (const float*)d_in[19]; const float* imb3 = (const float*)d_in[20];
    const float* imw4 = (const float*)d_in[21]; const float* imb4 = (const float*)d_in[22];
    float* out = (float*)d_out;

    // ---- workspace layout (256B aligned slabs) ----
    const size_t SZ_T1   = (size_t)B_*HP_*HP_*128*sizeof(_Float16); // 38,539,264
    const size_t SZ_FEAT = (size_t)B_*HP_*HP_*9*sizeof(_Float16);   //  2,709,792
    const size_t SZ_WCF  = (size_t)3*32*32*16*sizeof(_Float16);
    const size_t SZ_WMLP = (size_t)4*8*16*32*16*sizeof(_Float16);
    const size_t SZ_ACT  = (size_t)B_*Q_*HID_*sizeof(_Float16);     // 75,497,472
    auto al = [](size_t x){ return (x + 255) & ~(size_t)255; };
    char* ws = (char*)d_ws;
    _Float16* t1   = (_Float16*)(ws);
    _Float16* feat = (_Float16*)(ws + al(SZ_T1));
    _Float16* wcf  = (_Float16*)(ws + al(SZ_T1) + al(SZ_FEAT));
    _Float16* wmlp = (_Float16*)(ws + al(SZ_T1) + al(SZ_FEAT) + al(SZ_WCF));
    _Float16* act0 = (_Float16*)(ws + al(SZ_T1) + al(SZ_FEAT) + al(SZ_WCF) + al(SZ_WMLP));
    _Float16* act1 = (_Float16*)(ws + al(SZ_T1) + al(SZ_FEAT) + al(SZ_WCF) + al(SZ_WMLP) + al(SZ_ACT));

    // 1) zero padded t1+feat (borders must be zero for implicit 3x3 padding)
    long n16 = (long)((al(SZ_T1) + al(SZ_FEAT)) / 16);
    zero_kernel<<<(unsigned)((n16 + 255) / 256), 256, 0, stream>>>((uint4*)ws, n16);

    // 2) pack conv + MLP weights into WMMA B-fragment order (f16)
    int nprep = 3*32*32*16 + 4*8*16*32*16;
    prep_kernel<<<(nprep + 255) / 256, 256, 0, stream>>>(coefw, freqw, imw0, imw1, imw2, imw3,
                                                         wcf, wmlp);

    // 3) adapter: conv->sin (128ch), conv->sin (3ch), build feat [B,HP,HP,9] f16
    adapter1_kernel<<<B_*HW_, 128, 0, stream>>>(inp, a1w, a1b, t1);
    adapter2_kernel<<<(B_*HW_ + 255) / 256, 256, 0, stream>>>(inp, score, a2w, a2b, t1, feat);

    // 4) fused per-query conv-GEMM (WMMA) + sinusoidal encoding -> X = act0
    lte_query_kernel<<<B_*(Q_/16), 32, 0, stream>>>(coord, cell, coefb, freqb, phasew,
                                                    feat, wcf, act0);

    // 5) 4x ReLU MLP layers (WMMA GEMM, ping-pong)
    int mtiles = (B_*Q_) / 16;
    mlp_kernel<<<mtiles, 32, 0, stream>>>(act0, wmlp + 0*65536, imb0, act1);
    mlp_kernel<<<mtiles, 32, 0, stream>>>(act1, wmlp + 1*65536, imb1, act0);
    mlp_kernel<<<mtiles, 32, 0, stream>>>(act0, wmlp + 2*65536, imb2, act1);
    mlp_kernel<<<mtiles, 32, 0, stream>>>(act1, wmlp + 3*65536, imb3, act0);

    // 6) head 256->3 + bilinear-border residual
    final_kernel<<<(B_*Q_ + 255) / 256, 256, 0, stream>>>(act0, imw4, imb4, inp, coord, out);
    (void)in_sizes; (void)n_in; (void)out_size; (void)ws_size; (void)cell;
}
